// GCN_48876727828898
// MI455X (gfx1250) — matrix operand order repaired and checked by reading, not compiled, not run on hardware
//
#include <hip/hip_runtime.h>

// ---------------------------------------------------------------------------
// CDNA5 (gfx1250) implementation of the edge-GNN reference.
// Dense matmuls: V_WMMA_F32_16X16X32_BF16 (bf16 in, f32 accum).
// Flagship batched GEMM stages its shared B operand through LDS with
// GLOBAL_LOAD_ASYNC_TO_LDS_B128 double buffering (ASYNCcnt pipeline).
// ---------------------------------------------------------------------------

typedef __bf16 bf16_t;
typedef __attribute__((ext_vector_type(8)))  bf16_t v8bf;
typedef __attribute__((ext_vector_type(16))) bf16_t v16bf;
typedef __attribute__((ext_vector_type(8)))  float  v8f;

#define LRELU(v) ((v) >= 0.f ? (v) : 0.01f * (v))

static const int Bc = 16, Nn = 512, Mm = 1024, Hh = 128;

// ---------------- elementwise / helper kernels ----------------

__global__ void k_cvt_bf16(const float* __restrict__ in, bf16_t* __restrict__ out, int n) {
  int i = blockIdx.x * blockDim.x + threadIdx.x;
  if (i < n) out[i] = (bf16_t)in[i];
}

struct U4 { unsigned int a, b, c, d; };
__global__ void k_zero16(U4* __restrict__ p, long long n) {
  long long i = (long long)blockIdx.x * blockDim.x + threadIdx.x;
  if (i < n) { U4 z; z.a = z.b = z.c = z.d = 0u; p[i] = z; }
}

// W is [K x N] row-major f32; WT is [NPad x K] bf16 (zero padded rows)
__global__ void k_transpose_wt(const float* __restrict__ W, bf16_t* __restrict__ WT,
                               int K, int N, int NPad) {
  long long i = (long long)blockIdx.x * blockDim.x + threadIdx.x;
  if (i >= (long long)NPad * K) return;
  int k = (int)(i % K);
  int n = (int)(i / K);
  float v = (n < N) ? W[(size_t)k * N + n] : 0.f;
  WT[i] = (bf16_t)v;
}

// x = (nf0 - nf1 - bias)/scale ; layer0 of "initial": out = lrelu(x*W0 + b0)
__global__ void k_initial0(const float* __restrict__ nf, const float* __restrict__ sc,
                           const float* __restrict__ bs, const float* __restrict__ W0,
                           const float* __restrict__ b0, bf16_t* __restrict__ x0bf, int RN) {
  int i = blockIdx.x * blockDim.x + threadIdx.x;
  if (i >= RN * Hh) return;
  int h = i & (Hh - 1), r = i >> 7;
  float v = (nf[2 * r] - nf[2 * r + 1] - bs[0]) / sc[0];
  float o = v * W0[h] + b0[h];
  x0bf[i] = (bf16_t)LRELU(o);
}

// scale[b,m] = rsqrt(1 + act[b,m] * sum_e e2e[m,e]*act[b,e])
__global__ void k_scale(const float* __restrict__ e2e, const float* __restrict__ act,
                        float* __restrict__ scale) {
  int i = blockIdx.x * blockDim.x + threadIdx.x;
  if (i >= Bc * Mm) return;
  int m = i & (Mm - 1), b = i >> 10;
  const float* er = e2e + (size_t)m * Mm;
  const float* ar = act + (size_t)b * Mm;
  float s = 0.f;
  for (int e = 0; e < Mm; ++e) s += er[e] * ar[e];
  scale[i] = rsqrtf(1.f + ar[m] * s);
}

// A1hat[b,m,e] = scale_m*act_m * av1[b,k] * act_e*scale_e  at nonzeros of e2e
__global__ void k_scatter_a1(const int* __restrict__ idx, int nnz, int nnzp,
                             const float* __restrict__ av1, const float* __restrict__ act,
                             const float* __restrict__ scale, bf16_t* __restrict__ A1,
                             int total) {
  int i = blockIdx.x * blockDim.x + threadIdx.x;
  if (i >= total) return;
  int b = i & 15, k = i >> 4;
  int m = idx[k], e = idx[nnz + k];
  float v = av1[(size_t)b * nnzp + k] * act[b * Mm + m] * act[b * Mm + e] *
            scale[b * Mm + m] * scale[b * Mm + e];
  A1[(size_t)b * Mm * Mm + (size_t)m * Mm + e] = (bf16_t)v;
}

// ef0 = n2e_half @ x (2 nz per row); msg2 = (n2e_half*att2) @ x (loop invariant)
__global__ void k_ef0_msg2(const int* __restrict__ n2ei, int nnzn,
                           const float* __restrict__ n2e, const float* __restrict__ x,
                           const float* __restrict__ av2, float* __restrict__ ef,
                           bf16_t* __restrict__ efT, bf16_t* __restrict__ msg2bf) {
  int i = blockIdx.x * blockDim.x + threadIdx.x;
  if (i >= Bc * Mm * Hh) return;
  int h = i & (Hh - 1);
  int m = (i >> 7) & (Mm - 1);
  int b = i >> 17;
  int n1 = n2ei[nnzn + 2 * m], n2 = n2ei[nnzn + 2 * m + 1];
  float v1 = n2e[(size_t)m * Nn + n1], v2 = n2e[(size_t)m * Nn + n2];
  float x1 = x[((size_t)b * Nn + n1) * Hh + h];
  float x2 = x[((size_t)b * Nn + n2) * Hh + h];
  float e0 = v1 * x1 + v2 * x2;
  ef[i] = e0;
  efT[(size_t)b * Hh * Mm + (size_t)h * Mm + m] = (bf16_t)e0;
  float a1 = av2[b * nnzn + 2 * m], a2 = av2[b * nnzn + 2 * m + 1];
  msg2bf[i] = (bf16_t)(v1 * a1 * x1 + v2 * a2 * x2);
}

// ef = scale^2*ef + h1 + h2 ; refresh bf16 row-major + transposed copies
__global__ void k_update(float* __restrict__ ef, const float* __restrict__ h1,
                         const float* __restrict__ h2, const float* __restrict__ scale,
                         bf16_t* __restrict__ efT, bf16_t* __restrict__ efbf) {
  int i = blockIdx.x * blockDim.x + threadIdx.x;
  if (i >= Bc * Mm * Hh) return;
  int h = i & (Hh - 1);
  int m = (i >> 7) & (Mm - 1);
  int b = i >> 17;
  float s = scale[b * Mm + m];
  float v = s * s * ef[i] + h1[i] + h2[i];
  ef[i] = v;
  efbf[i] = (bf16_t)v;
  efT[(size_t)b * Hh * Mm + (size_t)h * Mm + m] = (bf16_t)v;
}

// final layer (64->3) + active-edge masking into d_out
__global__ void k_final2(const float* __restrict__ f1, const float* __restrict__ W2,
                         const float* __restrict__ b2, const float* __restrict__ act,
                         float* __restrict__ out) {
  int r = blockIdx.x * blockDim.x + threadIdx.x;
  if (r >= Bc * Mm) return;
  if (act[r] == 0.f) { out[3 * r] = 1.f; out[3 * r + 1] = 0.f; out[3 * r + 2] = 0.f; return; }
  const float* row = f1 + (size_t)r * 64;
#pragma unroll
  for (int j = 0; j < 3; ++j) {
    float s = b2[j];
    for (int k = 0; k < 64; ++k) s += row[k] * W2[k * 3 + j];
    out[3 * r + j] = s;
  }
}

// ---------------- generic bf16 WMMA GEMM ----------------
// C[R x Nout] = A[R x K] * BT^T  (+bias, optional lrelu). BT is [NoutPad x K].
// One wave per 16 x (16*NACC) tile. blockDim = (32, 4). Batched via blockIdx.z.
template <int NACC, int ACT>
__global__ __launch_bounds__(128) void k_gemm(
    const bf16_t* __restrict__ A, long long sA, const bf16_t* __restrict__ BT, long long sB,
    const float* __restrict__ bias, float* __restrict__ Cf, bf16_t* __restrict__ Cb,
    long long sC, int R, int K, int Nout, int ldc) {
  A += (long long)blockIdx.z * sA;
  BT += (long long)blockIdx.z * sB;
  if (Cf) Cf += (long long)blockIdx.z * sC;
  if (Cb) Cb += (long long)blockIdx.z * sC;

  int m0 = (blockIdx.x * blockDim.y + threadIdx.y) * 16;
  if (m0 >= R) return;
  int lane = threadIdx.x;
  int l15 = lane & 15;
  int kA = (lane >> 4) << 4;   // A frag: elems 0-15 -> K[kA..kA+15]
  int kB = (lane >> 4) << 4;   // B frag: elems 0-15 -> K[kB..kB+15]
  int n0 = blockIdx.y * (16 * NACC);
  const bf16_t* arow = A + (size_t)(m0 + l15) * K;

  v8f zero = {0.f, 0.f, 0.f, 0.f, 0.f, 0.f, 0.f, 0.f};
  v8f acc[NACC];
#pragma unroll
  for (int t = 0; t < NACC; ++t) acc[t] = zero;

  for (int k0 = 0; k0 < K; k0 += 32) {
    if (k0 + 32 < K) __builtin_prefetch(arow + k0 + 32, 0, 1);
    v8bf alo = *(const v8bf*)(arow + k0 + kA);
    v8bf ahi = *(const v8bf*)(arow + k0 + kA + 8);
    v16bf a = __builtin_shufflevector(alo, ahi, 0, 1, 2, 3, 4, 5, 6, 7,
                                      8, 9, 10, 11, 12, 13, 14, 15);
#pragma unroll
    for (int t = 0; t < NACC; ++t) {
      int col = n0 + t * 16 + l15;
      v16bf bfr = *(const v16bf*)(BT + (size_t)col * K + k0 + kB);
      acc[t] = __builtin_amdgcn_wmma_f32_16x16x32_bf16(false, a, false, bfr,
                                                       (short)0, acc[t], false, false);
    }
  }

  int rbase = m0 + ((lane >> 4) << 3);  // lanes 0-15: rows m0+0..7 ; 16-31: m0+8..15
#pragma unroll
  for (int t = 0; t < NACC; ++t) {
    int col = n0 + t * 16 + l15;
    if (col >= Nout) continue;
    float bv = bias ? bias[col] : 0.f;
#pragma unroll
    for (int j = 0; j < 8; ++j) {
      float v = acc[t][j] + bv;
      if (ACT) v = LRELU(v);
      size_t o = (size_t)(rbase + j) * ldc + col;
      if (Cf) Cf[o] = v;
      if (Cb) Cb[o] = (bf16_t)v;
    }
  }
}

// ---------------- flagship batched GEMM: msg1 = A1hat[b] @ ef[b] ----------------
// Per block: 64 M-rows x 128 H-cols of one batch. B operand (efT[b], shared by
// all 4 waves) is staged through LDS in 128x256 bf16 chunks with
// GLOBAL_LOAD_ASYNC_TO_LDS_B128, double-buffered on ASYNCcnt.
__global__ __launch_bounds__(128) void k_msg1(const bf16_t* __restrict__ A1,
                                              const bf16_t* __restrict__ Bt,
                                              bf16_t* __restrict__ out) {
  const int CH = 256;          // K elements per LDS chunk
  const int LDSP = CH + 8;     // padded row stride (16B) -> conflict-free columns
  __shared__ bf16_t sB[2][Hh * LDSP];

  const int lane = threadIdx.x, wy = threadIdx.y;
  const int tid = wy * 32 + lane;
  const int b = blockIdx.z;
  const int m0 = blockIdx.x * 64 + wy * 16;
  const int l15 = lane & 15;
  const int kA = (lane >> 4) << 4;
  const int kB = (lane >> 4) << 4;
  const bf16_t* Bb = Bt + (size_t)b * Hh * Mm;
  const bf16_t* arow = A1 + (size_t)b * Mm * Mm + (size_t)(m0 + l15) * Mm;

  v8f zero = {0.f, 0.f, 0.f, 0.f, 0.f, 0.f, 0.f, 0.f};
  v8f acc[8];
#pragma unroll
  for (int t = 0; t < 8; ++t) acc[t] = zero;

  // async-copy chunk c (efT[b][0:128][c*CH : c*CH+CH]) into sB[c&1]
  auto issue = [&](int c) {
    const int buf = c & 1;
#pragma unroll 4
    for (int s = 0; s < 32; ++s) {
      int g = tid + s * 128;        // 16B segment id, 32 segs per row
      int h = g >> 5;
      int kk = (g & 31) << 3;
      const bf16_t* src = Bb + (size_t)h * Mm + c * CH + kk;
      unsigned dst = (unsigned)(size_t)&sB[buf][h * LDSP + kk];
      asm volatile("global_load_async_to_lds_b128 %0, %1, off"
                   :: "v"(dst), "v"(src) : "memory");
    }
  };

  issue(0);
  for (int c = 0; c < Mm / CH; ++c) {
    if (c < Mm / CH - 1) {
      issue(c + 1);                                     // keep next chunk in flight
      asm volatile("s_wait_asynccnt 0x20" ::: "memory"); // chunk c complete
    } else {
      asm volatile("s_wait_asynccnt 0x0" ::: "memory");
    }
    __syncthreads();
    const bf16_t* sbuf = &sB[c & 1][0];
#pragma unroll
    for (int ks = 0; ks < CH; ks += 32) {
      const int k0 = c * CH + ks;
      v8bf alo = *(const v8bf*)(arow + k0 + kA);
      v8bf ahi = *(const v8bf*)(arow + k0 + kA + 8);
      v16bf a = __builtin_shufflevector(alo, ahi, 0, 1, 2, 3, 4, 5, 6, 7,
                                        8, 9, 10, 11, 12, 13, 14, 15);
#pragma unroll
      for (int t = 0; t < 8; ++t) {
        const bf16_t* bp = sbuf + (size_t)(t * 16 + l15) * LDSP + ks + kB;
        v8bf blo = *(const v8bf*)bp;
        v8bf bhi = *(const v8bf*)(bp + 8);
        v16bf bfr = __builtin_shufflevector(blo, bhi, 0, 1, 2, 3, 4, 5, 6, 7,
                                            8, 9, 10, 11, 12, 13, 14, 15);
        acc[t] = __builtin_amdgcn_wmma_f32_16x16x32_bf16(false, a, false, bfr,
                                                         (short)0, acc[t], false, false);
      }
    }
    __syncthreads();
  }

  const int rbase = m0 + ((lane >> 4) << 3);
  bf16_t* ob = out + (size_t)b * Mm * Hh;
#pragma unroll
  for (int t = 0; t < 8; ++t) {
    int col = t * 16 + l15;
#pragma unroll
    for (int j = 0; j < 8; ++j)
      ob[(size_t)(rbase + j) * Hh + col] = (bf16_t)acc[t][j];
  }
}

// ---------------- host orchestration ----------------

extern "C" void kernel_launch(void* const* d_in, const int* in_sizes, int n_in,
                              void* d_out, int out_size, void* d_ws, size_t ws_size,
                              hipStream_t stream) {
  (void)n_in; (void)out_size; (void)ws_size;
  const float* nf  = (const float*)d_in[0];
  const float* act = (const float*)d_in[1];
  const float* n2e = (const float*)d_in[2];
  const float* e2e = (const float*)d_in[3];
  const int* e2ei  = (const int*)d_in[4];
  const int* n2ei  = (const int*)d_in[5];
  const float* nsc = (const float*)d_in[6];
  const float* nbs = (const float*)d_in[7];
  const int NNZE = in_sizes[4] / 2;
  const int NNZEP = (NNZE + 15) & ~15;
  const int NNZN = in_sizes[5] / 2;  // 2048

  // params: attn1(8..15) attn2(16..23) initial(24..29) hid1(30..53) hid2(54..77) final(78..83)
  const float *a1W[4], *a1b[4], *a2W[4], *a2b[4], *iW[3], *ib[3], *fW[3], *fb[3];
  const float *h1W[4][3], *h1b[4][3], *h2W[4][3], *h2b[4][3];
  for (int l = 0; l < 4; ++l) { a1W[l] = (const float*)d_in[8 + 2 * l];  a1b[l] = (const float*)d_in[9 + 2 * l]; }
  for (int l = 0; l < 4; ++l) { a2W[l] = (const float*)d_in[16 + 2 * l]; a2b[l] = (const float*)d_in[17 + 2 * l]; }
  for (int l = 0; l < 3; ++l) { iW[l] = (const float*)d_in[24 + 2 * l];  ib[l] = (const float*)d_in[25 + 2 * l]; }
  for (int i = 0; i < 4; ++i)
    for (int l = 0; l < 3; ++l) {
      h1W[i][l] = (const float*)d_in[30 + i * 6 + 2 * l]; h1b[i][l] = (const float*)d_in[31 + i * 6 + 2 * l];
      h2W[i][l] = (const float*)d_in[54 + i * 6 + 2 * l]; h2b[i][l] = (const float*)d_in[55 + i * 6 + 2 * l];
    }
  for (int l = 0; l < 3; ++l) { fW[l] = (const float*)d_in[78 + 2 * l]; fb[l] = (const float*)d_in[79 + 2 * l]; }

  // workspace bump allocator (256B aligned)
  char* wp = (char*)d_ws; size_t off = 0;
  auto alloc = [&](size_t bytes) -> void* {
    void* p = wp + off; off += (bytes + 255) & ~(size_t)255; return p;
  };
  bf16_t* actbf = (bf16_t*)alloc((size_t)Bc * Mm * 2);
  bf16_t* WT0a = (bf16_t*)alloc(256 * 1024 * 2), *WT1a = (bf16_t*)alloc(256 * 256 * 2);
  bf16_t* WT2a = (bf16_t*)alloc(256 * 256 * 2),  *WT3a = (bf16_t*)alloc((size_t)NNZEP * 256 * 2);
  bf16_t* WT0b = (bf16_t*)alloc(256 * 1024 * 2), *WT1b = (bf16_t*)alloc(256 * 256 * 2);
  bf16_t* WT2b = (bf16_t*)alloc(256 * 256 * 2),  *WT3b = (bf16_t*)alloc((size_t)NNZN * 256 * 2);
  bf16_t* tA = (bf16_t*)alloc(16 * 256 * 2), *tB = (bf16_t*)alloc(16 * 256 * 2);
  float* av1 = (float*)alloc((size_t)Bc * NNZEP * 4);
  float* av2 = (float*)alloc((size_t)Bc * NNZN * 4);
  bf16_t* iWT1 = (bf16_t*)alloc(128 * 128 * 2), *iWT2 = (bf16_t*)alloc(128 * 128 * 2);
  bf16_t* x0bf = (bf16_t*)alloc((size_t)Bc * Nn * Hh * 2);
  bf16_t* x1bf = (bf16_t*)alloc((size_t)Bc * Nn * Hh * 2);
  float* xf = (float*)alloc((size_t)Bc * Nn * Hh * 4);
  float* scaleBuf = (float*)alloc((size_t)Bc * Mm * 4);
  bf16_t* A1hat = (bf16_t*)alloc((size_t)Bc * Mm * Mm * 2);
  float* ef = (float*)alloc((size_t)Bc * Mm * Hh * 4);
  bf16_t* efT = (bf16_t*)alloc((size_t)Bc * Hh * Mm * 2);
  bf16_t* efbf = (bf16_t*)alloc((size_t)Bc * Mm * Hh * 2);
  bf16_t* msg2bf = (bf16_t*)alloc((size_t)Bc * Mm * Hh * 2);
  bf16_t* msg1bf = (bf16_t*)alloc((size_t)Bc * Mm * Hh * 2);
  bf16_t *h1WT[4][3], *h2WT[4][3];
  for (int i = 0; i < 4; ++i)
    for (int l = 0; l < 3; ++l) {
      h1WT[i][l] = (bf16_t*)alloc(128 * 128 * 2);
      h2WT[i][l] = (bf16_t*)alloc(128 * 128 * 2);
    }
  bf16_t* u0bf = (bf16_t*)alloc((size_t)Bc * Mm * Hh * 2);
  bf16_t* u1bf = (bf16_t*)alloc((size_t)Bc * Mm * Hh * 2);
  float* h1f = (float*)alloc((size_t)Bc * Mm * Hh * 4);
  float* h2f = (float*)alloc((size_t)Bc * Mm * Hh * 4);
  bf16_t* fWT0 = (bf16_t*)alloc(128 * 128 * 2), *fWT1 = (bf16_t*)alloc(64 * 128 * 2);
  bf16_t* f0bf = (bf16_t*)alloc((size_t)Bc * Mm * Hh * 2);
  float* f1f = (float*)alloc((size_t)Bc * Mm * 64 * 4);

  auto T = [&](const float* W, bf16_t* WT, int K, int N, int NPad) {
    long long tot = (long long)NPad * K;
    k_transpose_wt<<<dim3((unsigned)((tot + 255) / 256)), dim3(256), 0, stream>>>(W, WT, K, N, NPad);
  };
  const dim3 gb(32, 4);

  // ---- precompute: weight transposes, bf16 input, degree scale ----
  k_cvt_bf16<<<dim3((Bc * Mm + 255) / 256), dim3(256), 0, stream>>>(act, actbf, Bc * Mm);
  T(a1W[0], WT0a, 1024, 256, 256); T(a1W[1], WT1a, 256, 256, 256);
  T(a1W[2], WT2a, 256, 256, 256);  T(a1W[3], WT3a, 256, NNZE, NNZEP);
  T(a2W[0], WT0b, 1024, 256, 256); T(a2W[1], WT1b, 256, 256, 256);
  T(a2W[2], WT2b, 256, 256, 256);  T(a2W[3], WT3b, 256, NNZN, NNZN);
  T(iW[1], iWT1, 128, 128, 128);   T(iW[2], iWT2, 128, 128, 128);
  for (int i = 0; i < 4; ++i)
    for (int l = 0; l < 3; ++l) { T(h1W[i][l], h1WT[i][l], 128, 128, 128); T(h2W[i][l], h2WT[i][l], 128, 128, 128); }
  T(fW[0], fWT0, 128, 128, 128);   T(fW[1], fWT1, 128, 64, 64);
  k_scale<<<dim3((Bc * Mm + 255) / 256), dim3(256), 0, stream>>>(e2e, act, scaleBuf);

  // ---- initial MLP: x = MLP([1,128,128,128]) ----
  k_initial0<<<dim3((Bc * Nn * Hh + 255) / 256), dim3(256), 0, stream>>>(nf, nsc, nbs, iW[0], ib[0], x0bf, Bc * Nn);
  k_gemm<8, 1><<<dim3(Bc * Nn / 64, 1, 1), gb, 0, stream>>>(x0bf, 0, iWT1, 0, ib[1], nullptr, x1bf, 0, Bc * Nn, 128, 128, 128);
  k_gemm<8, 0><<<dim3(Bc * Nn / 64, 1, 1), gb, 0, stream>>>(x1bf, 0, iWT2, 0, ib[2], xf, nullptr, 0, Bc * Nn, 128, 128, 128);

  // ---- attention MLPs (R=16 skinny WMMA GEMMs) ----
  k_gemm<1, 1><<<dim3(1, 16, 1), gb, 0, stream>>>(actbf, 0, WT0a, 0, a1b[0], nullptr, tA, 0, 16, 1024, 256, 256);
  k_gemm<1, 1><<<dim3(1, 16, 1), gb, 0, stream>>>(tA, 0, WT1a, 0, a1b[1], nullptr, tB, 0, 16, 256, 256, 256);
  k_gemm<1, 1><<<dim3(1, 16, 1), gb, 0, stream>>>(tB, 0, WT2a, 0, a1b[2], nullptr, tA, 0, 16, 256, 256, 256);
  k_gemm<1, 0><<<dim3(1, NNZEP / 16, 1), gb, 0, stream>>>(tA, 0, WT3a, 0, a1b[3], av1, nullptr, 0, 16, 256, NNZE, NNZEP);
  k_gemm<1, 1><<<dim3(1, 16, 1), gb, 0, stream>>>(actbf, 0, WT0b, 0, a2b[0], nullptr, tA, 0, 16, 1024, 256, 256);
  k_gemm<1, 1><<<dim3(1, 16, 1), gb, 0, stream>>>(tA, 0, WT1b, 0, a2b[1], nullptr, tB, 0, 16, 256, 256, 256);
  k_gemm<1, 1><<<dim3(1, 16, 1), gb, 0, stream>>>(tB, 0, WT2b, 0, a2b[2], nullptr, tA, 0, 16, 256, 256, 256);
  k_gemm<1, 0><<<dim3(1, NNZN / 16, 1), gb, 0, stream>>>(tA, 0, WT3b, 0, a2b[3], av2, nullptr, 0, 16, 256, NNZN, NNZN);

  // ---- A1hat (zero + scatter), ef0 and loop-invariant msg2 ----
  long long a1u4 = (long long)Bc * Mm * Mm * 2 / 16;
  k_zero16<<<dim3((unsigned)((a1u4 + 255) / 256)), dim3(256), 0, stream>>>((U4*)A1hat, a1u4);
  k_scatter_a1<<<dim3((NNZE * 16 + 255) / 256), dim3(256), 0, stream>>>(e2ei, NNZE, NNZEP, av1, act, scaleBuf, A1hat, NNZE * 16);
  k_ef0_msg2<<<dim3((Bc * Mm * Hh + 255) / 256), dim3(256), 0, stream>>>(n2ei, NNZN, n2e, xf, av2, ef, efT, msg2bf);

  // ---- 4 message-passing rounds ----
  for (int i = 0; i < 4; ++i) {
    // msg1 = A1hat @ ef : batched 1024x1024x128 bf16 WMMA GEMM (async-LDS staged B)
    k_msg1<<<dim3(Mm / 64, 1, Bc), gb, 0, stream>>>(A1hat, efT, msg1bf);
    // hid1 MLP(msg1)
    k_gemm<8, 1><<<dim3(Bc * Mm / 64, 1, 1), gb, 0, stream>>>(msg1bf, 0, h1WT[i][0], 0, h1b[i][0], nullptr, u0bf, 0, Bc * Mm, 128, 128, 128);
    k_gemm<8, 1><<<dim3(Bc * Mm / 64, 1, 1), gb, 0, stream>>>(u0bf, 0, h1WT[i][1], 0, h1b[i][1], nullptr, u1bf, 0, Bc * Mm, 128, 128, 128);
    k_gemm<8, 0><<<dim3(Bc * Mm / 64, 1, 1), gb, 0, stream>>>(u1bf, 0, h1WT[i][2], 0, h1b[i][2], h1f, nullptr, 0, Bc * Mm, 128, 128, 128);
    // hid2 MLP(msg2)
    k_gemm<8, 1><<<dim3(Bc * Mm / 64, 1, 1), gb, 0, stream>>>(msg2bf, 0, h2WT[i][0], 0, h2b[i][0], nullptr, u0bf, 0, Bc * Mm, 128, 128, 128);
    k_gemm<8, 1><<<dim3(Bc * Mm / 64, 1, 1), gb, 0, stream>>>(u0bf, 0, h2WT[i][1], 0, h2b[i][1], nullptr, u1bf, 0, Bc * Mm, 128, 128, 128);
    k_gemm<8, 0><<<dim3(Bc * Mm / 64, 1, 1), gb, 0, stream>>>(u1bf, 0, h2WT[i][2], 0, h2b[i][2], h2f, nullptr, 0, Bc * Mm, 128, 128, 128);
    // residual update (+ refresh bf16 copies)
    k_update<<<dim3((Bc * Mm * Hh + 255) / 256), dim3(256), 0, stream>>>(ef, h1f, h2f, scaleBuf, efT, efbf);
  }

  // ---- final head 128->128->64->3 + masking ----
  k_gemm<8, 1><<<dim3(Bc * Mm / 64, 1, 1), gb, 0, stream>>>(efbf, 0, fWT0, 0, fb[0], nullptr, f0bf, 0, Bc * Mm, 128, 128, 128);
  k_gemm<4, 1><<<dim3(Bc * Mm / 64, 1, 1), gb, 0, stream>>>(f0bf, 0, fWT1, 0, fb[1], f1f, nullptr, 0, Bc * Mm, 128, 64, 64);
  k_final2<<<dim3((Bc * Mm + 255) / 256), dim3(256), 0, stream>>>(f1f, fW[2], fb[2], act, (float*)d_out);
}
